// BitnetMultiGroupQueryAttention_41755672052100
// MI455X (gfx1250) — 16-bit and 8-bit matrix operand order repaired and checked by reading, compiled, not run on hardware
//
#include <hip/hip_runtime.h>
#include <cstdint>


#define BB   4
#define SS   2048
#define EE   2048
#define QH_  16
#define KH_  4
#define HD_  128
#define KVE_ 512
#define TT   (BB * SS)

typedef __attribute__((ext_vector_type(8)))  int      v8i;
typedef __attribute__((ext_vector_type(8)))  float    v8f;
typedef __attribute__((ext_vector_type(16))) _Float16 v16h;
typedef __attribute__((ext_vector_type(4)))  unsigned v4u;
typedef __attribute__((ext_vector_type(4)))  int      v4i;

#if defined(__has_builtin)
#  if __has_builtin(__builtin_amdgcn_tensor_load_to_lds)
#    define HAVE_TDM 1
#  endif
#endif
#ifndef HAVE_TDM
#  define HAVE_TDM 0
#endif


__global__ __launch_bounds__(256) void wabs_partial(const float* __restrict__ w,
                                                    int n, float* __restrict__ part) {
    __shared__ float red[256];
    float s = 0.f;
    for (int i = blockIdx.x * 256 + threadIdx.x; i < n; i += gridDim.x * 256)
        s += fabsf(w[i]);
    red[threadIdx.x] = s; __syncthreads();
    for (int o = 128; o; o >>= 1) {
        if (threadIdx.x < o) red[threadIdx.x] += red[threadIdx.x + o];
        __syncthreads();
    }
    if (threadIdx.x == 0) part[blockIdx.x] = red[0];
}

__global__ __launch_bounds__(256) void wscale_fin(const float* __restrict__ part,
                                                  float n, float* __restrict__ sc2) {
    __shared__ float red[256];
    red[threadIdx.x] = part[threadIdx.x]; __syncthreads();
    for (int o = 128; o; o >>= 1) {
        if (threadIdx.x < o) red[threadIdx.x] += red[threadIdx.x + o];
        __syncthreads();
    }
    if (threadIdx.x == 0) {
        float mean = fmaxf(red[0] / n, 1e-5f);
        sc2[0] = 1.f / mean;
        sc2[1] = mean;
    }
}

__global__ __launch_bounds__(256) void wquant(const float* __restrict__ w,
                                              const float* __restrict__ sc2,
                                              int8_t* __restrict__ w8, int n) {
    int i = blockIdx.x * 256 + threadIdx.x;
    if (i < n) {
        float q = rintf(w[i] * sc2[0]);
        w8[i] = (int8_t)fminf(fmaxf(q, -1.f), 1.f);
    }
}


__global__ __launch_bounds__(256) void act_quant_k(const float* __restrict__ X,
                                                   int8_t* __restrict__ x8,
                                                   float* __restrict__ deq, int K) {
    __shared__ float red[256];
    __shared__ float s_scale;
    int t = blockIdx.x, tid = threadIdx.x;
    const float* row = X + (size_t)t * K;
    float am = 0.f;
    for (int i = tid; i < K; i += 256) am = fmaxf(am, fabsf(row[i]));
    red[tid] = am; __syncthreads();
    for (int o = 128; o; o >>= 1) {
        if (tid < o) red[tid] = fmaxf(red[tid], red[tid + o]);
        __syncthreads();
    }
    if (tid == 0) {
        float a = fmaxf(red[0], 1e-5f);
        s_scale = 127.f / a;
        deq[t]  = a / 127.f;
    }
    __syncthreads();
    float sc = s_scale;
    for (int i = tid; i < K; i += 256)
        x8[(size_t)t * K + i] =
            (int8_t)fminf(fmaxf(rintf(row[i] * sc), -128.f), 127.f);
}


#if HAVE_TDM
__device__ __forceinline__ void tdm_load_2d(const void* gsrc, unsigned lds_off,
                                            unsigned stride_bytes,
                                            unsigned tile_w, unsigned tile_h) {
    unsigned long long ga = (unsigned long long)(uintptr_t)gsrc;
    v4u g0;
    g0[0] = 1u;
    g0[1] = lds_off;
    g0[2] = (unsigned)ga;
    g0[3] = (unsigned)((ga >> 32) & 0x1ffffffu)
            | (2u << 30);
    unsigned td0 = stride_bytes;
    unsigned td1 = 0x40000000u;
    v8i g1;
    g1[0] = 0;
    g1[1] = (int)((td0 & 0xffffu) << 16);
    g1[2] = (int)(((td0 >> 16) & 0xffffu) | ((td1 & 0xffffu) << 16));
    g1[3] = (int)(((td1 >> 16) & 0xffffu) | (tile_w << 16));
    g1[4] = (int)(tile_h & 0xffffu);
    g1[5] = (int)stride_bytes;
    g1[6] = 0;
    g1[7] = 0;
    v4i zz = {};
#if __clang_major__ >= 23
    v8i z8 = {};
    __builtin_amdgcn_tensor_load_to_lds(g0, g1, zz, zz, z8, 0);
#else
    __builtin_amdgcn_tensor_load_to_lds(g0, g1, zz, zz, 0);
#endif
}
#endif


__global__ __launch_bounds__(128) void gemm_i8(const int8_t* __restrict__ A,
                                               const int8_t* __restrict__ W,
                                               const float* __restrict__ adeq,
                                               const float* __restrict__ wsc2,
                                               const float* __restrict__ bias,
                                               void* __restrict__ Cout,
                                               int M, int N, int K, int mode) {
    __shared__ int lds_a[2][16 * 32];
    int tid  = threadIdx.x;
    int wave = tid >> 5;
    int lane = tid & 31;
    int nf   = lane & 15;
    int half = lane >> 4;
    int row0 = blockIdx.y * 16;
    int col0 = blockIdx.x * 64 + wave * 16;
    float wdeq = wsc2[1];
    const int8_t* Arow = A + (size_t)row0 * K;
    const int8_t* Wrow = W + (size_t)(col0 + nf) * K;

    v8i acc = {};
    const int nsteps = K >> 7;

#if HAVE_TDM
    bool issuer = (__builtin_amdgcn_readfirstlane(tid) == 0);
    unsigned lb0 = (unsigned)(uintptr_t)(void*)&lds_a[0][0];
    unsigned lb1 = (unsigned)(uintptr_t)(void*)&lds_a[1][0];
    if (issuer) {
        tdm_load_2d(Arow, lb0, (unsigned)K, 128u, 16u);
        __builtin_amdgcn_s_wait_tensorcnt(0);
    }
    __syncthreads();
    for (int it = 0; it < nsteps; ++it) {
        int cur = it & 1;
        if (issuer && (it + 1) < nsteps)
            tdm_load_2d(Arow + (size_t)(it + 1) * 128, (cur ? lb0 : lb1),
                        (unsigned)K, 128u, 16u);
        const int* la = &lds_a[cur][0];
        #pragma unroll
        for (int t = 0; t < 2; ++t) {
            v8i a, b;
            #pragma unroll
            for (int j = 0; j < 8; ++j) {
                int ka = ((j >> 2) << 5) + (((j >> 1) & 1) << 4) + ((j & 1) << 2) + (half << 3);
                a[j] = la[(nf << 5) + ((t * 64 + ka) >> 2)];
                int kb = ((j >> 2) << 5) + (((j >> 1) & 1) << 4) + ((j & 1) << 2) + (half << 3);
                b[j] = *(const int*)(Wrow + it * 128 + t * 64 + kb);
            }
            acc = __builtin_amdgcn_wmma_i32_16x16x64_iu8(true, a, true, b, acc, false, false);
        }
        if (issuer) __builtin_amdgcn_s_wait_tensorcnt(0);
        __syncthreads();
    }
#else
    for (int it = 0; it < nsteps; ++it) {
        int tmp[4];
        #pragma unroll
        for (int u = 0; u < 4; ++u) {
            int idx = tid + u * 128;
            int r = idx >> 5, c = idx & 31;
            tmp[u] = *(const int*)(Arow + (size_t)r * K + it * 128 + c * 4);
        }
        __syncthreads();
        #pragma unroll
        for (int u = 0; u < 4; ++u) lds_a[0][tid + u * 128] = tmp[u];
        __syncthreads();
        const int* la = &lds_a[0][0];
        #pragma unroll
        for (int t = 0; t < 2; ++t) {
            v8i a, b;
            #pragma unroll
            for (int j = 0; j < 8; ++j) {
                int ka = ((j >> 2) << 5) + (((j >> 1) & 1) << 4) + ((j & 1) << 2) + (half << 3);
                a[j] = la[(nf << 5) + ((t * 64 + ka) >> 2)];
            int kb = ((j >> 2) << 5) + (((j >> 1) & 1) << 4) + ((j & 1) << 2) + (half << 3);
                b[j] = *(const int*)(Wrow + it * 128 + t * 64 + kb);
            }
            acc = __builtin_amdgcn_wmma_i32_16x16x64_iu8(true, a, true, b, acc, false, false);
        }
    }
#endif

    #pragma unroll
    for (int r = 0; r < 8; ++r) {
        int row = row0 + r + (half << 3);
        int col = col0 + nf;
        float val = (float)acc[r] * adeq[row] * wdeq + bias[col];
        if (mode == 0) ((float*)Cout)[(size_t)row * N + col] = val;
        else ((_Float16*)Cout)[(size_t)row * N + col] = (_Float16)val;
    }
}


__global__ __launch_bounds__(256) void transpose_v(const _Float16* __restrict__ Vh,
                                                   _Float16* __restrict__ Vt) {
    size_t i = (size_t)blockIdx.x * 256 + threadIdx.x;
    int s  = (int)(i & (SS - 1));
    size_t j = i >> 11;
    int d  = (int)(j & (HD_ - 1));
    size_t l = j >> 7;
    int kh = (int)(l & (KH_ - 1));
    int b  = (int)(l >> 2);
    Vt[i] = Vh[((size_t)(b * SS + s)) * KVE_ + kh * HD_ + d];
}


union F16Vec { unsigned u[8]; v16h h; };

__global__ __launch_bounds__(32) void attn_flash(const float* __restrict__ Qf,
                                                 const _Float16* __restrict__ Kh,
                                                 const _Float16* __restrict__ Vt,
                                                 float* __restrict__ Xo) {
    __shared__ _Float16 ldsP[16 * 32];
    int lane = threadIdx.x;
    int nf = lane & 15, half = lane >> 4;
    int b = blockIdx.z, kh = blockIdx.y, nt = blockIdx.x;
    int n0 = nt * 16;
    const float inv_hd = 1.0f / (float)HD_;

    v16h qa[4];
    {
        const float* qrow = Qf + ((size_t)(b * SS + n0 + nf)) * EE + (kh * 4) * HD_;
        #pragma unroll
        for (int c = 0; c < 4; ++c) {
            #pragma unroll
            for (int j = 0; j < 8; ++j) {
                int k = c * 32 + ((j >> 2) << 4) + ((j & 3) << 1) + (half << 3);
                float v0 = qrow[k]     + qrow[HD_ + k]     + qrow[2 * HD_ + k]     + qrow[3 * HD_ + k];
                float v1 = qrow[k + 1] + qrow[HD_ + k + 1] + qrow[2 * HD_ + k + 1] + qrow[3 * HD_ + k + 1];
                qa[c][2 * j]     = (_Float16)(v0 * inv_hd);
                qa[c][2 * j + 1] = (_Float16)(v1 * inv_hd);
            }
        }
    }

    v8f oacc[8];
    float mrow[8], lrow[8];
    #pragma unroll
    for (int i = 0; i < 8; ++i) { oacc[i] = (v8f){}; mrow[i] = -1e30f; lrow[i] = 0.f; }

    int send = n0 + 16;
    int nsb  = (send + 31) >> 5;
    for (int sb = 0; sb < nsb; ++sb) {
        int s0 = sb << 5;
        v8f sc[2];
        #pragma unroll
        for (int st = 0; st < 2; ++st) {
            v8f s = (v8f){};
            int scol = s0 + st * 16 + nf;
            const _Float16* krow = Kh + ((size_t)(b * SS + scol)) * KVE_ + kh * HD_;
            #pragma unroll
            for (int c = 0; c < 4; ++c) {
                F16Vec kb;
                #pragma unroll
                for (int j = 0; j < 8; ++j) {
                    int d = c * 32 + ((j >> 2) << 4) + ((j & 3) << 1) + (half << 3);
                    kb.u[j] = *(const unsigned*)(krow + d);
                }
                s = __builtin_amdgcn_wmma_f32_16x16x32_f16(false, qa[c], false, kb.h,
                                                           (short)0, s, false, false);
            }
            #pragma unroll
            for (int r = 0; r < 8; ++r) {
                int n = n0 + r + (half << 3);
                if (scol > n) s[r] = -1e30f;
            }
            sc[st] = s;
        }

        float alpha[8];
        #pragma unroll
        for (int r = 0; r < 8; ++r) {
            float mx = fmaxf(sc[0][r], sc[1][r]);
            for (int off = 8; off; off >>= 1) mx = fmaxf(mx, __shfl_xor(mx, off, 32));
            float mnew = fmaxf(mrow[r], mx);
            float a  = __expf(mrow[r] - mnew);
            float p0 = __expf(sc[0][r] - mnew);
            float p1 = __expf(sc[1][r] - mnew);
            float rs = p0 + p1;
            for (int off = 8; off; off >>= 1) rs += __shfl_xor(rs, off, 32);
            lrow[r] = lrow[r] * a + rs;
            mrow[r] = mnew;
            alpha[r] = a;
            sc[0][r] = p0; sc[1][r] = p1;
        }

        __syncthreads();
        #pragma unroll
        for (int r = 0; r < 8; ++r) {
            int row = r + (half << 3);
            ldsP[row * 32 + nf]      = (_Float16)sc[0][r];
            ldsP[row * 32 + 16 + nf] = (_Float16)sc[1][r];
        }
        __syncthreads();
        v16h pa;
        #pragma unroll
        for (int j = 0; j < 8; ++j) {
            int k = ((j >> 2) << 4) + ((j & 3) << 1) + (half << 3);
            pa[2 * j]     = ldsP[nf * 32 + k];
            pa[2 * j + 1] = ldsP[nf * 32 + k + 1];
        }

        #pragma unroll
        for (int dt = 0; dt < 8; ++dt) {
            v8f o = oacc[dt];
            #pragma unroll
            for (int r = 0; r < 8; ++r) o[r] *= alpha[r];
            int d = dt * 16 + nf;
            const _Float16* vhead = Vt + (((size_t)(b * KH_ + kh) * HD_ + d) * SS) + s0;
            F16Vec vb;
            #pragma unroll
            for (int j = 0; j < 8; ++j) {
                int sl = ((j >> 2) << 4) + ((j & 3) << 1) + (half << 3);
                vb.u[j] = *(const unsigned*)(vhead + sl);
            }
            oacc[dt] = __builtin_amdgcn_wmma_f32_16x16x32_f16(false, pa, false, vb.h,
                                                              (short)0, o, false, false);
        }
    }

    #pragma unroll
    for (int dt = 0; dt < 8; ++dt)
        #pragma unroll
        for (int r = 0; r < 8; ++r) {
            int n = n0 + r + (half << 3);
            Xo[((size_t)(b * SS + n)) * KVE_ + kh * HD_ + dt * 16 + nf] =
                oacc[dt][r] / lrow[r];
        }
}


__global__ __launch_bounds__(256) void ln_quant(const float* __restrict__ X,
                                                const float* __restrict__ gamma,
                                                const float* __restrict__ beta,
                                                int8_t* __restrict__ x8,
                                                float* __restrict__ deq) {
    __shared__ float red[256];
    __shared__ float s_mean, s_rstd, s_scale;
    int t = blockIdx.x, tid = threadIdx.x;
    const float* row = X + (size_t)t * KVE_;
    float v0 = row[tid], v1 = row[tid + 256];

    red[tid] = v0 + v1; __syncthreads();
    for (int o = 128; o; o >>= 1) { if (tid < o) red[tid] += red[tid + o]; __syncthreads(); }
    if (tid == 0) s_mean = red[0] / (float)KVE_;
    __syncthreads();
    float mu = s_mean;
    float d0 = v0 - mu, d1 = v1 - mu;

    red[tid] = d0 * d0 + d1 * d1; __syncthreads();
    for (int o = 128; o; o >>= 1) { if (tid < o) red[tid] += red[tid + o]; __syncthreads(); }
    if (tid == 0) s_rstd = rsqrtf(red[0] / (float)KVE_ + 1e-5f);
    __syncthreads();
    float rstd = s_rstd;
    float y0 = d0 * rstd * gamma[tid]       + beta[tid];
    float y1 = d1 * rstd * gamma[tid + 256] + beta[tid + 256];

    red[tid] = fmaxf(fabsf(y0), fabsf(y1)); __syncthreads();
    for (int o = 128; o; o >>= 1) { if (tid < o) red[tid] = fmaxf(red[tid], red[tid + o]); __syncthreads(); }
    if (tid == 0) {
        float am = fmaxf(red[0], 1e-5f);
        s_scale = 127.f / am;
        deq[t]  = am / 127.f;
    }
    __syncthreads();
    float sc = s_scale;
    x8[(size_t)t * KVE_ + tid]       = (int8_t)fminf(fmaxf(rintf(y0 * sc), -128.f), 127.f);
    x8[(size_t)t * KVE_ + tid + 256] = (int8_t)fminf(fmaxf(rintf(y1 * sc), -128.f), 127.f);
}


extern "C" void kernel_launch(void* const* d_in, const int* in_sizes, int n_in,
                              void* d_out, int out_size, void* d_ws, size_t ws_size,
                              hipStream_t stream) {
    (void)in_sizes; (void)n_in; (void)out_size; (void)ws_size;

    const float* query = (const float*)d_in[0];
    const float* keyi  = (const float*)d_in[1];
    const float* vali  = (const float*)d_in[2];
    const float* Wq    = (const float*)d_in[3];
    const float* bq    = (const float*)d_in[4];
    const float* Wk    = (const float*)d_in[5];
    const float* bk    = (const float*)d_in[6];
    const float* Wv    = (const float*)d_in[7];
    const float* bv    = (const float*)d_in[8];
    const float* Wo    = (const float*)d_in[9];
    const float* bo    = (const float*)d_in[10];
    const float* gamma = (const float*)d_in[11];
    const float* beta  = (const float*)d_in[12];
    float* out = (float*)d_out;

    char* ws = (char*)d_ws;
    size_t off = 0;
    auto take = [&](size_t bytes) {
        size_t r = off;
        off = (off + bytes + 255) & ~(size_t)255;
        return r;
    };
    int8_t*   q8  = (int8_t*)(ws + take((size_t)TT * EE));
    int8_t*   k8  = (int8_t*)(ws + take((size_t)TT * EE));
    int8_t*   v8  = (int8_t*)(ws + take((size_t)TT * EE));
    int8_t*   x8  = (int8_t*)(ws + take((size_t)TT * KVE_));
    int8_t*   wq8 = (int8_t*)(ws + take((size_t)EE * EE));
    int8_t*   wk8 = (int8_t*)(ws + take((size_t)KVE_ * EE));
    int8_t*   wv8 = (int8_t*)(ws + take((size_t)KVE_ * EE));
    int8_t*   wo8 = (int8_t*)(ws + take((size_t)EE * KVE_));
    float*    qdq = (float*)(ws + take((size_t)TT * 4));
    float*    kdq = (float*)(ws + take((size_t)TT * 4));
    float*    vdq = (float*)(ws + take((size_t)TT * 4));
    float*    xdq = (float*)(ws + take((size_t)TT * 4));
    float*    wsc = (float*)(ws + take(8 * 4));
    float*    part = (float*)(ws + take(256 * 4));
    float*    Qf  = (float*)(ws + take((size_t)TT * EE * 4));
    _Float16* Khh = (_Float16*)(ws + take((size_t)TT * KVE_ * 2));
    _Float16* Vhh = (_Float16*)(ws + take((size_t)TT * KVE_ * 2));
    _Float16* Vt  = (_Float16*)(ws + take((size_t)TT * KVE_ * 2));
    float*    Xa  = (float*)(ws + take((size_t)TT * KVE_ * 4));

    struct WQ { const float* w; int n; int8_t* w8; float* sc; };
    WQ wjobs[4] = {
        { Wq, EE * EE,   wq8, wsc + 0 },
        { Wk, KVE_ * EE, wk8, wsc + 2 },
        { Wv, KVE_ * EE, wv8, wsc + 4 },
        { Wo, EE * KVE_, wo8, wsc + 6 },
    };
    for (int i = 0; i < 4; ++i) {
        wabs_partial<<<256, 256, 0, stream>>>(wjobs[i].w, wjobs[i].n, part);
        wscale_fin<<<1, 256, 0, stream>>>(part, (float)wjobs[i].n, wjobs[i].sc);
        wquant<<<(wjobs[i].n + 255) / 256, 256, 0, stream>>>(wjobs[i].w, wjobs[i].sc,
                                                             wjobs[i].w8, wjobs[i].n);
    }

    act_quant_k<<<TT, 256, 0, stream>>>(query, q8, qdq, EE);
    act_quant_k<<<TT, 256, 0, stream>>>(keyi,  k8, kdq, EE);
    act_quant_k<<<TT, 256, 0, stream>>>(vali,  v8, vdq, EE);

    gemm_i8<<<dim3(EE / 64,   TT / 16), 128, 0, stream>>>(q8, wq8, qdq, wsc + 0, bq, Qf,  TT, EE,   EE, 0);
    gemm_i8<<<dim3(KVE_ / 64, TT / 16), 128, 0, stream>>>(k8, wk8, kdq, wsc + 2, bk, Khh, TT, KVE_, EE, 1);
    gemm_i8<<<dim3(KVE_ / 64, TT / 16), 128, 0, stream>>>(v8, wv8, vdq, wsc + 4, bv, Vhh, TT, KVE_, EE, 1);

    transpose_v<<<(TT * KVE_) / 256, 256, 0, stream>>>(Vhh, Vt);

    attn_flash<<<dim3(SS / 16, KH_, BB), 32, 0, stream>>>(Qf, Khh, Vt, Xa);

    ln_quant<<<TT, 256, 0, stream>>>(Xa, gamma, beta, x8, xdq);
    gemm_i8<<<dim3(EE / 64, TT / 16), 128, 0, stream>>>(x8, wo8, xdq, wsc + 6, bo, out, TT, EE, KVE_, 0);
}
